// InterleavedLoRALinear_83949430767868
// MI455X (gfx1250) — matrix operand order repaired and checked by reading, not compiled, not run on hardware
//
#include <hip/hip_runtime.h>
#include <hip/hip_bf16.h>
#include <stddef.h>

// ---------------------------------------------------------------------------
// InterleavedLoRALinear for MI455X (gfx1250, wave32, WMMA).
//
// Math: all 4 adapters have alpha/r = 2.0, so
//   out = x @ (W + 2 * lora_b @ lora_a)^T = x @ W_eff^T
// Pass 1 builds W_eff in bf16 into d_ws (4096*4096*2 = 32 MB scratch).
// Pass 2 is a 8192x4096x4096 bf16 WMMA GEMM with f32 accumulation.
//   - 128x128 block tile, 4 wave32 waves, 64x64 per wave (4x4 WMMA tiles)
//   - KT=64 LDS stage, double buffered (72 KB LDS) -> 32 WMMAs per barrier
//   - W_eff tile staged with GLOBAL_LOAD_ASYNC_TO_LDS_B128 (ASYNCcnt) when
//     the toolchain exposes the builtin; X tile loaded f32 + cvt to bf16.
// ---------------------------------------------------------------------------

typedef __attribute__((ext_vector_type(4)))  float   vf4;
typedef __attribute__((ext_vector_type(8)))  float   v8f;
typedef __attribute__((ext_vector_type(4)))  __bf16  v4bf;
typedef __attribute__((ext_vector_type(8)))  __bf16  v8bf;
typedef __attribute__((ext_vector_type(16))) __bf16  v16bf;
typedef __attribute__((vector_size(16)))     int     v4i;     // b128 payload

typedef __attribute__((address_space(1))) v4i* gptr_v4i;      // global int4*
typedef __attribute__((address_space(3))) v4i* lptr_v4i;      // LDS int4*

union FragBF {           // 16 bf16 per lane = one WMMA A/B operand (8 VGPRs)
    v16bf v;
    v8bf  h[2];
};

#define IN_DIM   4096
#define OUT_DIM  4096
#define MROWS    8192          // BSZ * SEQ
#define RANK     64
#define BM       128
#define BN       128
#define KT       64
#define THREADS  128           // 4 waves of 32
#define PITCH    72            // LDS row pitch in bf16 elems (144B, 16B aligned)

#if __has_builtin(__builtin_amdgcn_global_load_async_to_lds_b128) && \
    __has_builtin(__builtin_amdgcn_s_wait_asynccnt)
#define USE_ASYNC 1
#else
#define USE_ASYNC 0
#endif

// ---------------------------------------------------------------------------
// Pass 1: W_eff[n,k] = W[n,k] + 2 * sum_r B[n,r] * A[r,k]   (bf16 output)
// grid: (IN_DIM/(256*4), OUT_DIM) = (4, 4096), block 256
// ---------------------------------------------------------------------------
__global__ __launch_bounds__(256)
void build_weff(const float* __restrict__ W,
                const float* __restrict__ A,
                const float* __restrict__ B,
                __bf16* __restrict__ Weff) {
    const int n  = blockIdx.y;
    const int k0 = (blockIdx.x * blockDim.x + threadIdx.x) * 4;

    __shared__ float c[RANK];
    if (threadIdx.x < RANK)
        c[threadIdx.x] = 2.0f * B[(size_t)n * RANK + threadIdx.x]; // alpha/r == 2
    __syncthreads();

    vf4 acc = *(const vf4*)(W + (size_t)n * IN_DIM + k0);
    #pragma unroll 8
    for (int r = 0; r < RANK; ++r) {
        vf4 a = *(const vf4*)(A + (size_t)r * IN_DIM + k0);
        acc += c[r] * a;
    }
    *(v4bf*)(Weff + (size_t)n * IN_DIM + k0) = __builtin_convertvector(acc, v4bf);
}

// ---------------------------------------------------------------------------
// Pass 2: Out[m,n] = sum_k Xbf[m,k] * Weff[n,k]   (f32 out)
// ---------------------------------------------------------------------------
__global__ __launch_bounds__(THREADS)
void lora_gemm(const float* __restrict__ X,
               const __bf16* __restrict__ Wf,
               float* __restrict__ Out) {
    __shared__ __align__(16) __bf16 As[2][BM * PITCH];
    __shared__ __align__(16) __bf16 Bs[2][BN * PITCH];

    const int t    = threadIdx.x;
    const int mBlk = blockIdx.y * BM;
    const int nBlk = blockIdx.x * BN;
    const int wave = t >> 5;
    const int lane = t & 31;
    const int wm   = (wave >> 1) * 64;     // wave row within block tile
    const int wn   = (wave & 1) * 64;      // wave col within block tile
    const int lr   = lane & 15;            // row/col within 16x16 tile
    const int lh   = lane >> 4;            // half-wave select

    // global -> LDS stage: W_eff via async-to-LDS copy, X via load+cvt
    auto loadStage = [&](int buf, int kBase) {
        const __bf16* wp = Wf + (size_t)(nBlk + t) * IN_DIM + kBase;
        __bf16* bs = &Bs[buf][t * PITCH];
#if USE_ASYNC
        #pragma unroll
        for (int i = 0; i < 8; ++i)
            __builtin_amdgcn_global_load_async_to_lds_b128(
                (gptr_v4i)(wp + 8 * i),
                (lptr_v4i)(bs + 8 * i),
                /*offset=*/0, /*cpol=*/0);
#else
        #pragma unroll
        for (int i = 0; i < 8; ++i)
            *(v8bf*)(bs + 8 * i) = *(const v8bf*)(wp + 8 * i);
#endif
        const float* xp = X + (size_t)(mBlk + t) * IN_DIM + kBase;
        if (kBase + KT < IN_DIM)
            __builtin_prefetch(xp + KT, 0, 0);   // global_prefetch next X stage
        __bf16* as = &As[buf][t * PITCH];
        #pragma unroll
        for (int i = 0; i < 16; ++i) {
            vf4 v = *(const vf4*)(xp + 4 * i);
            *(v4bf*)(as + 4 * i) = __builtin_convertvector(v, v4bf);
        }
    };

    v8f acc[4][4];
    #pragma unroll
    for (int i = 0; i < 4; ++i)
        #pragma unroll
        for (int j = 0; j < 4; ++j)
            acc[i][j] = (v8f)0.0f;

    const int nSteps = IN_DIM / KT;        // 64
    loadStage(0, 0);
#if USE_ASYNC
    __builtin_amdgcn_s_wait_asynccnt(0);   // async writes visible before barrier
#endif
    __syncthreads();

    for (int kt = 0; kt < nSteps; ++kt) {
        const int buf = kt & 1;
        if (kt + 1 < nSteps) loadStage(buf ^ 1, (kt + 1) * KT);

        #pragma unroll
        for (int ks = 0; ks < KT; ks += 32) {
            // A frags: CDNA5 16-bit A 16x32 layout:
            //   lanes 0-15 hold K{0..7,16..23}, lanes 16-31 hold K{8..15,24..31}
            FragBF a[4], b[4];
            #pragma unroll
            for (int i = 0; i < 4; ++i) {
                const __bf16* p = &As[buf][(wm + i * 16 + lr) * PITCH + ks];
                a[i].h[0] = *(const v8bf*)(p + lh * 8);
                a[i].h[1] = *(const v8bf*)(p + 16 + lh * 8);
            }
            // B frags: 32x16 layout: lanes 0-15 K{0..7,16..23}, lanes 16-31 K{8..15,24..31}
            #pragma unroll
            for (int j = 0; j < 4; ++j) {
                const __bf16* p = &Bs[buf][(wn + j * 16 + lr) * PITCH + ks];
                b[j].h[0] = *(const v8bf*)(p + lh * 8);
                b[j].h[1] = *(const v8bf*)(p + 16 + lh * 8);
            }

            #pragma unroll
            for (int i = 0; i < 4; ++i)
                #pragma unroll
                for (int j = 0; j < 4; ++j)
                    acc[i][j] = __builtin_amdgcn_wmma_f32_16x16x32_bf16(
                        /*neg_a=*/false, a[i].v,
                        /*neg_b=*/false, b[j].v,
                        /*c_mod=*/(short)0, acc[i][j],
                        /*reuse_a=*/false, /*reuse_b=*/false);
        }

#if USE_ASYNC
        __builtin_amdgcn_s_wait_asynccnt(0);
#endif
        __syncthreads();
    }

    // D layout: VGPR g holds M = g + 8*(lane/16), N = lane%16
    #pragma unroll
    for (int i = 0; i < 4; ++i) {
        #pragma unroll
        for (int j = 0; j < 4; ++j) {
            const int mB = mBlk + wm + i * 16 + 8 * lh;
            const int nB = nBlk + wn + j * 16 + lr;
            #pragma unroll
            for (int g = 0; g < 8; ++g)
                Out[(size_t)(mB + g) * OUT_DIM + nB] = acc[i][j][g];
        }
    }
}

// ---------------------------------------------------------------------------
extern "C" void kernel_launch(void* const* d_in, const int* in_sizes, int n_in,
                              void* d_out, int out_size, void* d_ws, size_t ws_size,
                              hipStream_t stream) {
    (void)in_sizes; (void)n_in; (void)out_size; (void)ws_size;
    const float* x  = (const float*)d_in[0];   // [4,2048,4096]
    const float* W  = (const float*)d_in[1];   // [4096,4096]
    const float* La = (const float*)d_in[2];   // [64,4096]
    const float* Lb = (const float*)d_in[3];   // [4096,64]
    float* out = (float*)d_out;                // [4,2048,4096]
    __bf16* Weff = (__bf16*)d_ws;              // 32 MB scratch

    build_weff<<<dim3(IN_DIM / (256 * 4), OUT_DIM), 256, 0, stream>>>(W, La, Lb, Weff);
    lora_gemm<<<dim3(OUT_DIM / BN, MROWS / BM), THREADS, 0, stream>>>(x, Weff, out);
}
